// Affine_Linear_X_YZ_22067541967106
// MI455X (gfx1250) — matrix operand order repaired and checked by reading, not compiled, not run on hardware
//
#include <hip/hip_runtime.h>

typedef __attribute__((ext_vector_type(16))) _Float16 v16h;
typedef __attribute__((ext_vector_type(8)))  _Float16 v8h;
typedef __attribute__((ext_vector_type(8)))  float    v8f;
typedef __attribute__((ext_vector_type(4)))  float    v4f;

#define NPTS   32              // (b,n) points per workgroup
#define NCOL   96              // NPTS * 3 columns
#define TPITCH 72              // f16 pitch per column (64 + 8 pad -> conflict-free, 16B aligned)
#define OPITCH 68              // f32 pitch per column for output staging

__global__ __launch_bounds__(128)
void affine_linear_wmma(const float* __restrict__ X, const float* __restrict__ J,
                        const float* __restrict__ A, const float* __restrict__ Bm,
                        const float* __restrict__ C, float* __restrict__ Y)
{
    // Ta/Tc (a_term/c_term, f16) alias the fp32 output staging buffer (dead after GEMM1).
    __shared__ union SU {
        _Float16 T[2][NCOL * TPITCH];    // 2 * 13824 B
        float    Yout[NCOL * OPITCH];    // 26112 B
    } su;
    __shared__ _Float16 Y1h[NCOL * TPITCH]; // intermediate Y1 in f16

    const int  t    = threadIdx.x;
    const int  lane = t & 31;
    const int  wave = t >> 5;      // row tile of f (0..3)
    const int  l16  = lane & 15;
    const int  lhi  = lane >> 4;   // 0/1
    const long pbase = (long)blockIdx.x * NPTS;

    // ---------------- Phase 1: geometry (fp32 VALU) -> f16 tiles in LDS ----------------
    for (int it = 0; it < 16; ++it) {
        int  idx = it * 128 + t;           // 0..2047 = 32 points x 64 channels
        int  p   = idx >> 6;
        int  e   = idx & 63;
        long gp  = pbase + p;
        const float* jp = J + gp * 384 + e * 6;   // (3,2) row-major per channel
        const float* xp = X + gp * 192 + e * 3;
        float a1x = jp[0], a1y = jp[2], a1z = jp[4];
        float a2x = jp[1], a2y = jp[3], a2z = jp[5];
        float x0 = xp[0], x1 = xp[1], x2 = xp[2];

        float inv1 = 1.0f / fmaxf(sqrtf(a1x*a1x + a1y*a1y + a1z*a1z), 1e-12f);
        float b1x = a1x*inv1, b1y = a1y*inv1, b1z = a1z*inv1;
        float d   = b1x*a2x + b1y*a2y + b1z*a2z;
        float ux  = a2x - d*b1x, uy = a2y - d*b1y, uz = a2z - d*b1z;
        float inv2 = 1.0f / fmaxf(sqrtf(ux*ux + uy*uy + uz*uz), 1e-12f);
        float b2x = ux*inv2, b2y = uy*inv2, b2z = uz*inv2;
        float b3x = b1y*b2z - b1z*b2y;
        float b3y = b1z*b2x - b1x*b2z;
        float b3z = b1x*b2y - b1y*b2x;

        float rt0 = b1x*x0 + b1y*x1 + b1z*x2;
        float rt1 = b2x*x0 + b2y*x1 + b2z*x2;
        float rt2 = b3x*x0 + b3y*x1 + b3z*x2;
        float s = rt1 - rt2, q = rt1 + rt2;

        int c0 = p * 3;
        su.T[0][(c0+0)*TPITCH + e] = (_Float16)(b2x*s + b3x*q);
        su.T[0][(c0+1)*TPITCH + e] = (_Float16)(b2y*s + b3y*q);
        su.T[0][(c0+2)*TPITCH + e] = (_Float16)(b2z*s + b3z*q);
        su.T[1][(c0+0)*TPITCH + e] = (_Float16)(b1x*rt0);
        su.T[1][(c0+1)*TPITCH + e] = (_Float16)(b1y*rt0);
        su.T[1][(c0+2)*TPITCH + e] = (_Float16)(b1z*rt0);
    }
    __syncthreads();

    // A-fragment of a 64x64 weight: lane m = rowtile*16 + l16; elem j <-> k = kt*32 + 8*lhi + 16*(j>>3) + (j&7)
    auto load_w = [&](const float* __restrict__ W, int kt) -> v16h {
        const float* wp = W + (wave*16 + l16)*64 + kt*32 + 8*lhi;
        v4f w0 = *(const v4f*)(wp +  0);
        v4f w1 = *(const v4f*)(wp +  4);
        v4f w2 = *(const v4f*)(wp + 16);
        v4f w3 = *(const v4f*)(wp + 20);
        v16h f;
#pragma unroll
        for (int j = 0; j < 4; ++j) {
            f[j]      = (_Float16)w0[j];
            f[j + 4]  = (_Float16)w1[j];
            f[j + 8]  = (_Float16)w2[j];
            f[j + 12] = (_Float16)w3[j];
        }
        return f;
    };

    // B-fragment from LDS tile [col][e]: lane n = ct*16 + l16; elem j <-> k = kt*32 + 8*lhi + 16*(j>>3) + (j&7)
    auto load_b = [&](const _Float16* __restrict__ T, int ct, int kt) -> v16h {
        const _Float16* p = T + (ct*16 + l16)*TPITCH + kt*32 + 8*lhi;
        v8h lo = *(const v8h*)(p);
        v8h hi = *(const v8h*)(p + 16);
        return __builtin_shufflevector(lo, hi, 0,1,2,3,4,5,6,7,8,9,10,11,12,13,14,15);
    };

    v16h Af0 = load_w(A,  0), Af1 = load_w(A,  1);
    v16h Cf0 = load_w(C,  0), Cf1 = load_w(C,  1);
    v16h Bf0 = load_w(Bm, 0), Bf1 = load_w(Bm, 1);

    // ---------------- GEMM1: Y1 = A @ Ta + C @ Tc  (f32 accum, stored f16) ----------------
#pragma unroll
    for (int ct = 0; ct < 6; ++ct) {
        v16h ta0 = load_b(su.T[0], ct, 0), ta1 = load_b(su.T[0], ct, 1);
        v16h tc0 = load_b(su.T[1], ct, 0), tc1 = load_b(su.T[1], ct, 1);
        v8f acc = {};
        acc = __builtin_amdgcn_wmma_f32_16x16x32_f16(false, Af0, false, ta0, (short)0, acc, false, false);
        acc = __builtin_amdgcn_wmma_f32_16x16x32_f16(false, Af1, false, ta1, (short)0, acc, false, false);
        acc = __builtin_amdgcn_wmma_f32_16x16x32_f16(false, Cf0, false, tc0, (short)0, acc, false, false);
        acc = __builtin_amdgcn_wmma_f32_16x16x32_f16(false, Cf1, false, tc1, (short)0, acc, false, false);
        v8h h;
#pragma unroll
        for (int r = 0; r < 8; ++r) h[r] = (_Float16)acc[r];
        *(v8h*)&Y1h[(ct*16 + l16)*TPITCH + wave*16 + 8*lhi] = h;   // D elem r -> f = rowtile*16 + 8*lhi + r
    }
    __syncthreads();

    // ---------------- GEMM2: Y = Bm @ Y1 -> fp32 staging (aliases Ta/Tc) ----------------
#pragma unroll
    for (int ct = 0; ct < 6; ++ct) {
        v16h y0 = load_b(Y1h, ct, 0), y1 = load_b(Y1h, ct, 1);
        v8f acc = {};
        acc = __builtin_amdgcn_wmma_f32_16x16x32_f16(false, Bf0, false, y0, (short)0, acc, false, false);
        acc = __builtin_amdgcn_wmma_f32_16x16x32_f16(false, Bf1, false, y1, (short)0, acc, false, false);
        float* op = &su.Yout[(ct*16 + l16)*OPITCH + wave*16 + 8*lhi];
        *(v4f*)(op)     = (v4f){acc[0], acc[1], acc[2], acc[3]};
        *(v4f*)(op + 4) = (v4f){acc[4], acc[5], acc[6], acc[7]};
    }
    __syncthreads();

    // ---------------- Coalesced float4 writeback: out[p][f][i] <- Yout[p*3+i][f] ----------------
    const long obase = (long)blockIdx.x * (NPTS * 192);
#pragma unroll
    for (int it = 0; it < 12; ++it) {
        int o = (it * 128 + t) * 4;     // 6144 floats per block
        v4f v;
#pragma unroll
        for (int u = 0; u < 4; ++u) {
            int oo  = o + u;
            int p   = oo / 192;
            int rem = oo - p * 192;
            int f   = rem / 3;
            int i   = rem - f * 3;
            v[u] = su.Yout[(p*3 + i)*OPITCH + f];
        }
        *(v4f*)(Y + obase + o) = v;
    }
}

extern "C" void kernel_launch(void* const* d_in, const int* in_sizes, int n_in,
                              void* d_out, int out_size, void* d_ws, size_t ws_size,
                              hipStream_t stream) {
    const float* X  = (const float*)d_in[0];
    const float* J  = (const float*)d_in[1];
    const float* A  = (const float*)d_in[2];
    const float* Bm = (const float*)d_in[3];
    const float* C  = (const float*)d_in[4];
    float* Y = (float*)d_out;

    int npoints = in_sizes[0] / 192;       // B*N = 65536
    int nblocks = npoints / NPTS;          // 2048
    affine_linear_wmma<<<nblocks, 128, 0, stream>>>(X, J, A, Bm, C, Y);
}
